// MultiHeadAttention_67130338836499
// MI455X (gfx1250) — matrix operand order repaired and checked by reading, not compiled, not run on hardware
//
#include <hip/hip_runtime.h>
#include <hip/hip_bf16.h>
#include <stdint.h>

#define D_MODEL 1024
#define HEADS   16
#define KD      64
#define SEQ     2048
#define BATCH   4
#define MROWS   (BATCH * SEQ)   // 8192

typedef __attribute__((ext_vector_type(16))) __bf16 v16bf;
typedef __attribute__((ext_vector_type(8)))  float  v8f;

union ABFrag { uint4 q[2]; v16bf v; };

__device__ __forceinline__ unsigned short f2bf(float f) {
  union { float f; unsigned int u; } c; c.f = f;
  unsigned int u = c.u;
  return (unsigned short)((u + 0x7FFFu + ((u >> 16) & 1u)) >> 16);  // RNE
}

__device__ __forceinline__ v8f vzero8() {
  v8f v;
#pragma unroll
  for (int i = 0; i < 8; ++i) v[i] = 0.0f;
  return v;
}

// ---------------------------------------------------------------------------
// One wave computes a 32(M) x 64(N) f32 tile of A[M,K] @ Bt[N,K]^T with
// v_wmma_f32_16x16x32_bf16, K unrolled by 64 with ping-pong fragment buffers
// so next chunk's global loads overlap with current chunk's WMMAs.
// A-fragment: lane = row (l16), halves hold K 0..15 / 16..31 contiguously.
// B-fragment: lane = column (l16), halves hold K 0..15 / 16..31 contiguously.
// ---------------------------------------------------------------------------
__device__ __forceinline__ void gemm_tile_32x64(const unsigned short* __restrict__ A,
                                                const unsigned short* __restrict__ Bt,
                                                int K, int mbase, int nbase, int lane,
                                                v8f acc[2][4]) {
  const int half = lane >> 4;
  const int l16  = lane & 15;
  const unsigned short* arow[2];
  const unsigned short* brow[4];
#pragma unroll
  for (int i = 0; i < 2; ++i)
    arow[i] = A + (size_t)(mbase + i * 16 + l16) * K;
#pragma unroll
  for (int j = 0; j < 4; ++j)
    brow[j] = Bt + (size_t)(nbase + j * 16 + l16) * K + half * 16;

  auto loadA = [&](int kc, ABFrag a[2]) {
#pragma unroll
    for (int i = 0; i < 2; ++i) {
      a[i].q[0] = *(const uint4*)(arow[i] + kc + half * 16);
      a[i].q[1] = *(const uint4*)(arow[i] + kc + half * 16 + 8);
    }
  };
  auto loadB = [&](int kc, ABFrag b[4]) {
#pragma unroll
    for (int j = 0; j < 4; ++j) {
      const unsigned short* p = brow[j] + kc;
      b[j].q[0] = ((const uint4*)p)[0];
      b[j].q[1] = ((const uint4*)p)[1];
    }
  };
  auto mma = [&](ABFrag a[2], ABFrag b[4]) {
#pragma unroll
    for (int i = 0; i < 2; ++i)
#pragma unroll
      for (int j = 0; j < 4; ++j)
        acc[i][j] = __builtin_amdgcn_wmma_f32_16x16x32_bf16(
            false, a[i].v, false, b[j].v, (short)0, acc[i][j], false, false);
  };

  ABFrag a0[2], b0[4], a1[2], b1[4];
  loadA(0, a0); loadB(0, b0);
  for (int kc = 0; kc < K; kc += 64) {
    loadA(kc + 32, a1); loadB(kc + 32, b1);   // prefetch 2nd half-chunk
    mma(a0, b0);
    if (kc + 64 < K) { loadA(kc + 64, a0); loadB(kc + 64, b0); }  // next iter
    mma(a1, b1);
  }
}

// ------------------------- conversion kernels ------------------------------
__global__ __launch_bounds__(256) void cvt_f32_bf16(const float* __restrict__ in,
                                                    unsigned short* __restrict__ out,
                                                    int n) {
  int i = blockIdx.x * 256 + threadIdx.x;
  if (i < n) out[i] = f2bf(in[i]);
}

// W[k][n] (K x N, row major) -> Wt[n][k] bf16
__global__ __launch_bounds__(256) void transpose_to_bf16(const float* __restrict__ W,
                                                         unsigned short* __restrict__ Wt,
                                                         int K, int N) {
  long i = (long)blockIdx.x * 256 + threadIdx.x;
  if (i < (long)K * N) {
    int n = (int)(i / K);
    int k = (int)(i % K);
    Wt[i] = f2bf(W[(size_t)k * N + n]);
  }
}

// ------------------------------ QKV GEMM -----------------------------------
// qkv[m, n] = x[m,:] @ W_attn[:, n] + b_attn[n];  reference split order is
// (value, query, key).  present = d_out tail: [2][B][H][S][KD] with k first.
__global__ __launch_bounds__(256) void qkv_gemm(const unsigned short* __restrict__ xb,
                                                const unsigned short* __restrict__ Wat,
                                                const float* __restrict__ b_attn,
                                                float* __restrict__ present,
                                                unsigned short* __restrict__ qws,
                                                unsigned short* __restrict__ kws,
                                                unsigned short* __restrict__ vtws) {
  const int lane = threadIdx.x & 31;
  const int wid  = threadIdx.x >> 5;
  const int mbase = blockIdx.x * 64 + (wid >> 2) * 32;
  const int nbase = blockIdx.y * 256 + (wid & 3) * 64;
  const int half = lane >> 4, l16 = lane & 15;

  v8f acc[2][4];
#pragma unroll
  for (int i = 0; i < 2; ++i)
#pragma unroll
    for (int j = 0; j < 4; ++j) acc[i][j] = vzero8();

  gemm_tile_32x64(xb, Wat, D_MODEL, mbase, nbase, lane, acc);

#pragma unroll
  for (int j = 0; j < 4; ++j) {
    const int n = nbase + j * 16 + l16;
    const int sec = n >> 10;          // 0=value 1=query 2=key
    const int nn = n & 1023;
    const int h = nn >> 6, d = nn & 63;
    const float bias = b_attn[n];
#pragma unroll
    for (int i = 0; i < 2; ++i)
#pragma unroll
      for (int r = 0; r < 8; ++r) {
        const int m = mbase + i * 16 + r + 8 * half;
        const int b = m >> 11;          // /SEQ
        const int s = m & (SEQ - 1);
        const float val = acc[i][j][r] + bias;
        if (sec == 0) {                 // value -> present[1], v^T bf16
          present[((((size_t)BATCH + b) * HEADS + h) * SEQ + s) * KD + d] = val;
          vtws[(((size_t)(b * HEADS + h)) * KD + d) * SEQ + s] = f2bf(val);
        } else if (sec == 1) {          // query bf16 [b,h,s,d]
          qws[(((size_t)(b * HEADS + h)) * SEQ + s) * KD + d] = f2bf(val);
        } else {                        // key -> present[0], k bf16 [b,h,s,d]
          present[(((size_t)b * HEADS + h) * SEQ + s) * KD + d] = val;
          kws[(((size_t)(b * HEADS + h)) * SEQ + s) * KD + d] = f2bf(val);
        }
      }
  }
}

// ------------------------- flash attention ---------------------------------
// Block = 8 consecutive 16-row query tiles of one (b,h).  K/V 64-key chunks
// are staged in LDS once per block via async global->LDS copies (ASYNCcnt,
// double buffered), then all waves consume them; online softmax per wave.
__global__ __launch_bounds__(256) void attn_kernel(const unsigned short* __restrict__ qws,
                                                   const unsigned short* __restrict__ kws,
                                                   const unsigned short* __restrict__ vtws,
                                                   unsigned short* __restrict__ ctx) {
  __shared__ alignas(16) unsigned short kbuf[2][64][64];  // [buf][s][d]   16 KB
  __shared__ alignas(16) unsigned short vbuf[2][64][64];  // [buf][d][s]   16 KB
  __shared__ alignas(16) unsigned short plds[8][16 * 64]; // per-wave P    16 KB

  const int lane = threadIdx.x & 31;
  const int wid  = threadIdx.x >> 5;
  const int idx  = blockIdx.x * 8 + wid;     // [0, B*H*S/16)
  const int qt0 = (blockIdx.x * 8) & 127;    // first q-tile in block (same b,h)
  const int qt = idx & 127;                  // S/16 = 128
  const int h  = (idx >> 7) & 15;
  const int b  = idx >> 11;
  const int qbase = qt * 16;
  const int half = lane >> 4, l16 = lane & 15;

  const unsigned short* qrow =
      qws + ((size_t)(b * HEADS + h) * SEQ + qbase + l16) * KD;
  ABFrag qf[2];
#pragma unroll
  for (int c = 0; c < 2; ++c) {
    qf[c].q[0] = *(const uint4*)(qrow + c * 32 + half * 16);
    qf[c].q[1] = *(const uint4*)(qrow + c * 32 + half * 16 + 8);
  }

  const unsigned short* kbp = kws  + (size_t)(b * HEADS + h) * SEQ * KD;
  const unsigned short* vbp = vtws + (size_t)(b * HEADS + h) * KD * SEQ;

  v8f acc[4];
#pragma unroll
  for (int j = 0; j < 4; ++j) acc[j] = vzero8();
  float mrow[8], lrow[8], scl[8];
#pragma unroll
  for (int r = 0; r < 8; ++r) { mrow[r] = -1e30f; lrow[r] = 0.0f; }

  // async stage one 64-key K tile + V tile into LDS buffer `bufi`
  // (each thread moves 2x16B of K and 2x16B of V -> 4 async instrs per wave)
  const int srow = (int)(threadIdx.x >> 3);        // 0..31
  const int sseg = ((int)threadIdx.x & 7) * 8;     // ushort offset, 16B step
  auto issue = [&](int kb2, int bufi) {
#pragma unroll
    for (int p = 0; p < 2; ++p) {
      const int rr = srow + p * 32;                // 0..63
      unsigned long long gk = (unsigned long long)(uintptr_t)
          (kbp + (size_t)(kb2 + rr) * KD + sseg);
      unsigned lk = (unsigned)(uintptr_t)&kbuf[bufi][rr][sseg];
      asm volatile("global_load_async_to_lds_b128 %0, %1, off"
                   :: "v"(lk), "v"(gk) : "memory");
      unsigned long long gv = (unsigned long long)(uintptr_t)
          (vbp + (size_t)rr * SEQ + kb2 + sseg);
      unsigned lv = (unsigned)(uintptr_t)&vbuf[bufi][rr][sseg];
      asm volatile("global_load_async_to_lds_b128 %0, %1, off"
                   :: "v"(lv), "v"(gv) : "memory");
    }
  };

  const int kmax_w = qbase + 16;                   // this wave's causal limit
  const int kmaxb  = qt0 * 16 + 128;               // block limit (last wave)

  issue(0, 0);
  for (int kb = 0; kb < kmaxb; kb += 64) {
    const int buf = (kb >> 6) & 1;
    const bool more = (kb + 64 < kmaxb);
    __syncthreads();                               // release buf^1 (prev chunk)
    if (more) issue(kb + 64, buf ^ 1);
    if (more) asm volatile("s_wait_asynccnt 0x4" ::: "memory");
    else      asm volatile("s_wait_asynccnt 0x0" ::: "memory");
    __syncthreads();                               // buf data ready block-wide

    if (kb < kmax_w) {                             // wave-uniform guard
      // ---- scores: four 16x16 tiles, K=64 contraction, frags from LDS ----
      v8f sc[4];
#pragma unroll
      for (int t = 0; t < 4; ++t) {
        v8f s = vzero8();
#pragma unroll
        for (int c = 0; c < 2; ++c) {
          ABFrag kf;
          const unsigned short* kr = &kbuf[buf][t * 16 + l16][c * 32 + half * 16];
          kf.q[0] = ((const uint4*)kr)[0];
          kf.q[1] = ((const uint4*)kr)[1];
          s = __builtin_amdgcn_wmma_f32_16x16x32_bf16(
              false, qf[c].v, false, kf.v, (short)0, s, false, false);
        }
        sc[t] = s;
      }

      // ---- mask + scale + online softmax (per C-fragment row) ----
#pragma unroll
      for (int r = 0; r < 8; ++r) {
        const int row = qbase + r + 8 * half;
        float sv[4];
#pragma unroll
        for (int t = 0; t < 4; ++t) {
          const int col = kb + t * 16 + l16;
          sv[t] = (col <= row) ? sc[t][r] * 0.125f : -1e10f;
        }
        float mx = fmaxf(fmaxf(sv[0], sv[1]), fmaxf(sv[2], sv[3]));
        mx = fmaxf(mx, __shfl_xor(mx, 1, 32));
        mx = fmaxf(mx, __shfl_xor(mx, 2, 32));
        mx = fmaxf(mx, __shfl_xor(mx, 4, 32));
        mx = fmaxf(mx, __shfl_xor(mx, 8, 32));
        const float mn = fmaxf(mrow[r], mx);
        const float sc_ = __expf(mrow[r] - mn);
        mrow[r] = mn;
        scl[r] = sc_;
        float p[4], sum = 0.0f;
#pragma unroll
        for (int t = 0; t < 4; ++t) { p[t] = __expf(sv[t] - mn); sum += p[t]; }
        sum += __shfl_xor(sum, 1, 32);
        sum += __shfl_xor(sum, 2, 32);
        sum += __shfl_xor(sum, 4, 32);
        sum += __shfl_xor(sum, 8, 32);
        lrow[r] = lrow[r] * sc_ + sum;
        const int rl = r + 8 * half;
#pragma unroll
        for (int t = 0; t < 4; ++t)
          plds[wid][rl * 64 + t * 16 + l16] = f2bf(p[t]);
      }

      // rescale running context
#pragma unroll
      for (int j = 0; j < 4; ++j)
#pragma unroll
        for (int r = 0; r < 8; ++r) acc[j][r] *= scl[r];

      // wave-local LDS relayout: C-frag -> A-frag (DS in-order per wave)
      asm volatile("s_wait_dscnt 0" ::: "memory");
      ABFrag pf[2];
      const unsigned short* prow = &plds[wid][l16 * 64];
#pragma unroll
      for (int c = 0; c < 2; ++c) {
        pf[c].q[0] = *(const uint4*)(prow + c * 32 + half * 16);
        pf[c].q[1] = *(const uint4*)(prow + c * 32 + half * 16 + 8);
      }

      // ---- context += P(16x64) @ V(64x64); V tile in LDS as [d][s] ----
#pragma unroll
      for (int j = 0; j < 4; ++j) {
#pragma unroll
        for (int c = 0; c < 2; ++c) {
          ABFrag vf;
          const unsigned short* vr = &vbuf[buf][j * 16 + l16][c * 32 + half * 16];
          vf.q[0] = ((const uint4*)vr)[0];
          vf.q[1] = ((const uint4*)vr)[1];
          acc[j] = __builtin_amdgcn_wmma_f32_16x16x32_bf16(
              false, pf[c].v, false, vf.v, (short)0, acc[j], false, false);
        }
      }
    }
  }

  // ---- normalize, merge heads, emit bf16 context row for proj GEMM ----
#pragma unroll
  for (int j = 0; j < 4; ++j)
#pragma unroll
    for (int r = 0; r < 8; ++r) {
      const int row = qbase + r + 8 * half;
      const float o = acc[j][r] / lrow[r];
      ctx[((size_t)(b * SEQ + row)) * D_MODEL + h * KD + j * 16 + l16] = f2bf(o);
    }
}

// ------------------------------ proj GEMM ----------------------------------
__global__ __launch_bounds__(256) void proj_gemm(const unsigned short* __restrict__ ctx,
                                                 const unsigned short* __restrict__ Wpt,
                                                 const float* __restrict__ b_proj,
                                                 float* __restrict__ out) {
  const int lane = threadIdx.x & 31;
  const int wid  = threadIdx.x >> 5;
  const int mbase = blockIdx.x * 64 + (wid >> 2) * 32;
  const int nbase = blockIdx.y * 256 + (wid & 3) * 64;
  const int half = lane >> 4, l16 = lane & 15;

  v8f acc[2][4];
#pragma unroll
  for (int i = 0; i < 2; ++i)
#pragma unroll
    for (int j = 0; j < 4; ++j) acc[i][j] = vzero8();

  gemm_tile_32x64(ctx, Wpt, D_MODEL, mbase, nbase, lane, acc);

#pragma unroll
  for (int j = 0; j < 4; ++j) {
    const int n = nbase + j * 16 + l16;
    const float bias = b_proj[n];
#pragma unroll
    for (int i = 0; i < 2; ++i)
#pragma unroll
      for (int r = 0; r < 8; ++r) {
        const int m = mbase + i * 16 + r + 8 * half;
        out[(size_t)m * D_MODEL + n] = acc[i][j][r] + bias;
      }
  }
}

// ---------------------------------------------------------------------------
extern "C" void kernel_launch(void* const* d_in, const int* in_sizes, int n_in,
                              void* d_out, int out_size, void* d_ws, size_t ws_size,
                              hipStream_t stream) {
  (void)in_sizes; (void)n_in; (void)out_size; (void)ws_size;
  const float* x      = (const float*)d_in[0];
  const float* W_attn = (const float*)d_in[1];
  const float* b_attn = (const float*)d_in[2];
  const float* W_proj = (const float*)d_in[3];
  const float* b_proj = (const float*)d_in[4];

  float* out     = (float*)d_out;
  float* present = out + (size_t)MROWS * D_MODEL;   // [2,B,H,S,KD]

  // workspace partition (bf16 halves), ~88 MB total
  unsigned short* xb   = (unsigned short*)d_ws;
  unsigned short* Wat  = xb   + (size_t)MROWS * D_MODEL;           // x bf16
  unsigned short* Wpt  = Wat  + (size_t)D_MODEL * 3 * D_MODEL;     // W_attn^T
  unsigned short* qws  = Wpt  + (size_t)D_MODEL * D_MODEL;         // W_proj^T
  unsigned short* kws  = qws  + (size_t)BATCH * HEADS * SEQ * KD;
  unsigned short* vtws = kws  + (size_t)BATCH * HEADS * SEQ * KD;
  unsigned short* ctx  = vtws + (size_t)BATCH * HEADS * SEQ * KD;

  cvt_f32_bf16<<<(MROWS * D_MODEL) / 256, 256, 0, stream>>>(x, xb, MROWS * D_MODEL);
  transpose_to_bf16<<<(3 * D_MODEL * D_MODEL + 255) / 256, 256, 0, stream>>>(
      W_attn, Wat, D_MODEL, 3 * D_MODEL);
  transpose_to_bf16<<<(D_MODEL * D_MODEL + 255) / 256, 256, 0, stream>>>(
      W_proj, Wpt, D_MODEL, D_MODEL);

  qkv_gemm<<<dim3(MROWS / 64, (3 * D_MODEL) / 256), 256, 0, stream>>>(
      xb, Wat, b_attn, present, qws, kws, vtws);

  attn_kernel<<<(BATCH * HEADS * (SEQ / 16)) / 8, 256, 0, stream>>>(
      qws, kws, vtws, ctx);

  proj_gemm<<<dim3(MROWS / 64, D_MODEL / 256), 256, 0, stream>>>(
      ctx, Wpt, b_proj, out);
}
